// NetworkLocal_23733989278191
// MI455X (gfx1250) — matrix operand order repaired and checked by reading, not compiled, not run on hardware
//
#include <hip/hip_runtime.h>
#include <hip/hip_bf16.h>

// ---- types -----------------------------------------------------------------
typedef __attribute__((ext_vector_type(16))) _Float16 v16h;
typedef __attribute__((ext_vector_type(8)))  float    v8f;
typedef __attribute__((ext_vector_type(4)))  unsigned int u32x4; // POD 128-bit

union V16 { v16h v; u32x4 q[2]; };

__device__ __forceinline__ v8f wmma_f16(const V16& a, const V16& b, v8f c) {
  // 8 args: (neg_a, A, neg_b, B, c_mod, C, reuse_a, reuse_b)
  return __builtin_amdgcn_wmma_f32_16x16x32_f16(false, a.v, false, b.v,
                                                (short)0, c, false, false);
}

// leaky_relu(x), slope 0.01:  leaky(x) = 0.505*x + 0.495*|x|  (exact identity)
__device__ __forceinline__ float leaky(float x) {
  return fmaf(0.495f, fabsf(x), 0.505f * x);
}

// sum_r leaky(c[r]) = 0.505*sum(c) + 0.495*sum(|c|)  -- |x| is a free src modifier
__device__ __forceinline__ float pool_leaky8(const v8f& c) {
  float s0 = (c[0] + c[1]) + (c[2] + c[3]);
  float s1 = (c[4] + c[5]) + (c[6] + c[7]);
  float a0 = (fabsf(c[0]) + fabsf(c[1])) + (fabsf(c[2]) + fabsf(c[3]));
  float a1 = (fabsf(c[4]) + fabsf(c[5])) + (fabsf(c[6]) + fabsf(c[7]));
  return fmaf(0.495f, a0 + a1, 0.505f * (s0 + s1));
}

// ---- problem constants -----------------------------------------------------
#define BZ   4
#define NTT  1024
#define NDM  128
#define KIN  19
#define HH   256
#define LL   128
#define EPSF 1e-5f

// workspace byte offsets (16B aligned)
#define WS_W0T   0              // 256*32  f16 = 16384 B   (W0^T padded K->32)
#define WS_W1T   16384          // 256*256 f16 = 131072 B  (W1^T)
#define WS_W2T   147456         // 128*256 f16 = 65536 B   (W2^T)
#define WS_HP    212992         // 4096*256 f16 = 2097152 B (pooled hidden)
// total ws needed: 2310144 bytes (~2.3 MB)

// ---- kernel 0: convert/trans/pad weights to f16 ----------------------------
__global__ __launch_bounds__(256) void prep_weights(
    const float* __restrict__ W0, const float* __restrict__ W1,
    const float* __restrict__ W2, _Float16* __restrict__ w0t,
    _Float16* __restrict__ w1t, _Float16* __restrict__ w2t) {
  int tid = blockIdx.x * 256 + threadIdx.x;
  const int N0 = 256 * 32;            // W0T elements
  const int N1 = 256 * 256;           // W1T elements
  const int N2 = 128 * 256;           // W2T elements
  if (tid < N0) {
    int n = tid >> 5, k = tid & 31;
    float val = (k < KIN) ? W0[k * HH + n] : 0.0f;   // W0 [19][256] -> [256][32]
    w0t[tid] = (_Float16)val;
  } else if (tid < N0 + N1) {
    int i = tid - N0; int n = i >> 8, k = i & 255;   // W1 [256][256] -> [256][256]^T
    w1t[i] = (_Float16)W1[k * HH + n];
  } else if (tid < N0 + N1 + N2) {
    int i = tid - (N0 + N1); int n = i >> 8, k = i & 255; // W2 [256][128] -> [128][256]^T
    w2t[i] = (_Float16)W2[k * LL + n];
  }
}

// ---- kernel 1: features + layer0 GEMM (WMMA) + mean pool -------------------
__global__ __launch_bounds__(256) void feat_mlp0(
    const float* __restrict__ x0, const float* __restrict__ x,
    const int* __restrict__ Nin, const float* __restrict__ basis,
    const float* __restrict__ v, const _Float16* __restrict__ w0t,
    const float* __restrict__ b0, _Float16* __restrict__ hpool) {
  __shared__ __align__(16) _Float16 sA[NDM * 32];   // features, f16, K padded to 32
  __shared__ float pool[HH];
  __shared__ float b0s[HH];
  __shared__ float vsum[3];
  __shared__ float x0s[3];
  __shared__ float bas[9];
  __shared__ int   Nsh;

  const int tid = threadIdx.x;
  const int bt  = blockIdx.x;       // b*NT + t
  const int bi  = bt >> 10;         // batch index

  // warm L2/WGP$ with the W0^T table (16 KB = 256 threads * 64B lines)
  __builtin_prefetch((const char*)w0t + tid * 64, 0, 0);

  if (tid < 3) vsum[tid] = 0.0f;
  pool[tid] = 0.0f;                                  // blockDim == 256 == HH
  b0s[tid]  = b0[tid];
  if (tid < 3) x0s[tid] = x0[(size_t)bt * 3 + tid];
  if (tid < 9) bas[tid] = basis[(size_t)bi * 9 + tid];
  if (tid == 0) Nsh = Nin[bt];
  __syncthreads();

  float vdx = 0.f, vdy = 0.f, vdz = 0.f, xdx = 0.f, xdy = 0.f, xdz = 0.f;
  if (tid < NDM) {
    const float* xp = x + ((size_t)bt * NDM + tid) * 3;
    const float* vp = v + ((size_t)bt * NDM + tid) * 3;
    xdx = xp[0]; xdy = xp[1]; xdz = xp[2];
    vdx = vp[0]; vdy = vp[1]; vdz = vp[2];
    atomicAdd(&vsum[0], vdx);                        // ds_add_f32
    atomicAdd(&vsum[1], vdy);
    atomicAdd(&vsum[2], vdz);
  }
  __syncthreads();

  if (tid < NDM) {
    const float inv = 1.0f / (float)NDM;
    float v0x = vsum[0] * inv, v0y = vsum[1] * inv, v0z = vsum[2] * inv;
    float f[KIN];
    f[0] = log1pf((float)Nsh);
    float ax = x0s[0], ay = x0s[1], az = x0s[2];
    float x0n = sqrtf(ax*ax + ay*ay + az*az) + EPSF;
    f[1] = x0n;
    float x0ux = ax/x0n, x0uy = ay/x0n, x0uz = az/x0n;
    f[2] = x0ux*bas[0] + x0uy*bas[1] + x0uz*bas[2];
    f[3] = x0ux*bas[3] + x0uy*bas[4] + x0uz*bas[5];
    f[4] = x0ux*bas[6] + x0uy*bas[7] + x0uz*bas[8];
    float v0n = sqrtf(v0x*v0x + v0y*v0y + v0z*v0z) + EPSF;
    f[5] = v0n;
    float v0ux = v0x/v0n, v0uy = v0y/v0n, v0uz = v0z/v0n;
    f[6] = v0ux*bas[0] + v0uy*bas[1] + v0uz*bas[2];
    f[7] = v0ux*bas[3] + v0uy*bas[4] + v0uz*bas[5];
    f[8] = v0ux*bas[6] + v0uy*bas[7] + v0uz*bas[8];
    f[9] = x0ux*v0ux + x0uy*v0uy + x0uz*v0uz;
    float xn = sqrtf(xdx*xdx + xdy*xdy + xdz*xdz) + EPSF;
    f[10] = xn;
    float xux = xdx/xn, xuy = xdy/xn, xuz = xdz/xn;
    f[11] = xux*bas[0] + xuy*bas[1] + xuz*bas[2];
    f[12] = xux*bas[3] + xuy*bas[4] + xuz*bas[5];
    f[13] = xux*bas[6] + xuy*bas[7] + xuz*bas[8];
    float wx = vdx - v0x, wy = vdy - v0y, wz = vdz - v0z;
    float vn = sqrtf(wx*wx + wy*wy + wz*wz) + EPSF;
    f[14] = vn;
    float vux = wx/vn, vuy = wy/vn, vuz = wz/vn;
    f[15] = vux*bas[0] + vuy*bas[1] + vuz*bas[2];
    f[16] = vux*bas[3] + vuy*bas[4] + vuz*bas[5];
    f[17] = vux*bas[6] + vuy*bas[7] + vuz*bas[8];
    f[18] = xux*vux + xuy*vuy + xuz*vuz;

    _Float16* row = sA + tid * 32;
    #pragma unroll
    for (int k = 0; k < KIN; ++k) row[k] = (_Float16)f[k];
    #pragma unroll
    for (int k = KIN; k < 32; ++k) row[k] = (_Float16)0.0f;
  }
  __syncthreads();

  // ---- WMMA phase: wave w handles rows 16w..16w+15, all 16 N-tiles,
  //      two tiles per iteration so epilogue VALU hides the WMMA->VALU hazard
  const int wv = tid >> 5, lane = tid & 31;
  const int lg = lane >> 4, ln = lane & 15;
  const int arow = wv * 16 + ln;
  const u32x4* sA4 = (const u32x4*)sA;               // 8 halves per u32x4

  V16 A;                                             // A: K = 8g+0..7 | 16+8g+0..7
  A.q[0] = sA4[arow * 4 + lg];
  A.q[1] = sA4[arow * 4 + 2 + lg];

  const u32x4* w0t4 = (const u32x4*)w0t;
  #pragma unroll
  for (int nc = 0; nc < 16; nc += 2) {
    const int n0 = nc * 16 + ln;
    const int n1 = n0 + 16;
    V16 B0, B1;                                      // B: K = 8g+0..7 | 16+8g+0..7
    B0.q[0] = w0t4[n0 * 4 + lg];
    B0.q[1] = w0t4[n0 * 4 + 2 + lg];
    B1.q[0] = w0t4[n1 * 4 + lg];
    B1.q[1] = w0t4[n1 * 4 + 2 + lg];
    const float bias0 = b0s[n0];
    const float bias1 = b0s[n1];
    v8f c0, c1;
    #pragma unroll
    for (int r = 0; r < 8; ++r) { c0[r] = bias0; c1[r] = bias1; }
    c0 = wmma_f16(A, B0, c0);                        // back-to-back WMMAs:
    c1 = wmma_f16(A, B1, c1);                        // no D->A/B overlap
    atomicAdd(&pool[n0], pool_leaky8(c0));           // epilogue0 fills c1 hazard
    atomicAdd(&pool[n1], pool_leaky8(c1));
  }
  __syncthreads();

  hpool[(size_t)bt * HH + tid] = (_Float16)(pool[tid] * (1.0f / (float)NDM));
}

// ---- kernel 2: layers 1 and 2 on pooled [4096,256] -------------------------
__global__ __launch_bounds__(128) void mlp12(
    const _Float16* __restrict__ hpool, const _Float16* __restrict__ w1t,
    const float* __restrict__ b1, const _Float16* __restrict__ w2t,
    const float* __restrict__ b2, float* __restrict__ out) {
  __shared__ __align__(16) _Float16 h1[16 * 264];    // layer-1 output tile, padded stride

  const int tid = threadIdx.x;
  const int wv = tid >> 5, lane = tid & 31;
  const int lg = lane >> 4, ln = lane & 15;
  const int m0 = blockIdx.x * 16;

  // warm caches with both weight tables
  __builtin_prefetch((const char*)w1t + tid * 64, 0, 0);
  __builtin_prefetch((const char*)w2t + tid * 64, 0, 0);

  const u32x4* hp4  = (const u32x4*)hpool;
  const u32x4* w1t4 = (const u32x4*)w1t;
  const u32x4* w2t4 = (const u32x4*)w2t;

  // ---- layer 1: K=256, each wave owns 4 N-tiles; bias preloaded into C ----
  v8f acc[4];
  #pragma unroll
  for (int i = 0; i < 4; ++i) {
    const float bb = b1[(wv + 4 * i) * 16 + ln];
    v8f c;
    #pragma unroll
    for (int r = 0; r < 8; ++r) c[r] = bb;
    acc[i] = c;
  }
  const int arow = m0 + ln;
  #pragma unroll
  for (int kc = 0; kc < 8; ++kc) {
    V16 A;
    A.q[0] = hp4[arow * 32 + kc * 4 + lg];
    A.q[1] = hp4[arow * 32 + kc * 4 + 2 + lg];
    #pragma unroll
    for (int i = 0; i < 4; ++i) {
      const int n = (wv + 4 * i) * 16 + ln;
      V16 Bm;
      Bm.q[0] = w1t4[n * 32 + kc * 4 + lg];
      Bm.q[1] = w1t4[n * 32 + kc * 4 + 2 + lg];
      acc[i] = wmma_f16(A, Bm, acc[i]);
    }
  }
  // leaky-relu, stash as f16 in LDS for layer 2's A operand
  #pragma unroll
  for (int i = 0; i < 4; ++i) {
    const int n = (wv + 4 * i) * 16 + ln;
    #pragma unroll
    for (int r = 0; r < 8; ++r)
      h1[(r + 8 * lg) * 264 + n] = (_Float16)leaky(acc[i][r]); // C/D: M = r + 8g
  }
  __syncthreads();

  // ---- layer 2: K=256, N=128; each wave owns 2 N-tiles ----
  const u32x4* h14 = (const u32x4*)h1;               // row stride 264 halves = 33 u32x4
  v8f acc2[2];
  #pragma unroll
  for (int i = 0; i < 2; ++i) {
    const float bb = b2[(wv + 4 * i) * 16 + ln];
    v8f c;
    #pragma unroll
    for (int r = 0; r < 8; ++r) c[r] = bb;
    acc2[i] = c;
  }
  #pragma unroll
  for (int kc = 0; kc < 8; ++kc) {
    V16 A;
    A.q[0] = h14[ln * 33 + kc * 4 + lg];
    A.q[1] = h14[ln * 33 + kc * 4 + 2 + lg];
    #pragma unroll
    for (int i = 0; i < 2; ++i) {
      const int n = (wv + 4 * i) * 16 + ln;
      V16 Bm;
      Bm.q[0] = w2t4[n * 32 + kc * 4 + lg];
      Bm.q[1] = w2t4[n * 32 + kc * 4 + 2 + lg];
      acc2[i] = wmma_f16(A, Bm, acc2[i]);
    }
  }
  #pragma unroll
  for (int i = 0; i < 2; ++i) {
    const int n = (wv + 4 * i) * 16 + ln;
    #pragma unroll
    for (int r = 0; r < 8; ++r)
      out[(size_t)(m0 + r + 8 * lg) * LL + n] = acc2[i][r];
  }
}

// ---- host launch -----------------------------------------------------------
extern "C" void kernel_launch(void* const* d_in, const int* in_sizes, int n_in,
                              void* d_out, int out_size, void* d_ws, size_t ws_size,
                              hipStream_t stream) {
  const float* x0    = (const float*)d_in[0];
  const float* x     = (const float*)d_in[1];
  const int*   Nin   = (const int*)d_in[2];
  const float* basis = (const float*)d_in[3];
  const float* v     = (const float*)d_in[4];
  const float* W0    = (const float*)d_in[5];
  const float* b0    = (const float*)d_in[6];
  const float* W1    = (const float*)d_in[7];
  const float* b1    = (const float*)d_in[8];
  const float* W2    = (const float*)d_in[9];
  const float* b2    = (const float*)d_in[10];
  float* out = (float*)d_out;
  char* ws = (char*)d_ws;

  _Float16* w0t   = (_Float16*)(ws + WS_W0T);
  _Float16* w1t   = (_Float16*)(ws + WS_W1T);
  _Float16* w2t   = (_Float16*)(ws + WS_W2T);
  _Float16* hpool = (_Float16*)(ws + WS_HP);

  // 256*32 + 256*256 + 128*256 = 106496 weight elements
  prep_weights<<<(106496 + 255) / 256, 256, 0, stream>>>(W0, W1, W2, w0t, w1t, w2t);
  feat_mlp0<<<BZ * NTT, 256, 0, stream>>>(x0, x, Nin, basis, v, w0t, b0, hpool);
  mlp12<<<(BZ * NTT) / 16, 128, 0, stream>>>(hpool, w1t, b1, w2t, b2, out);
}
